// InputsTreeEmbedding_38886633898793
// MI455X (gfx1250) — matrix operand order repaired and checked by reading, not compiled, not run on hardware
//
#include <hip/hip_runtime.h>
#include <hip/hip_bf16.h>

// Problem constants (match reference)
#define V_SZ   100000
#define D_SZ   256
#define N2_SZ  1000
#define N3_SZ  100
#define B_SZ   64
#define S_SZ   512
#define NTOK   (B_SZ * S_SZ)     // 32768 tokens
#define GATES  1024              // 4*D
#define KCAT   512               // [x | h] concatenated K

typedef __attribute__((ext_vector_type(16))) __bf16 v16bf;
typedef __attribute__((ext_vector_type(8)))  float  v8f;

union FragBF { v16bf bf; unsigned u[8]; };

__device__ __forceinline__ unsigned short f2bf(float f) {
  unsigned x = __float_as_uint(f);
  x += 0x7FFFu + ((x >> 16) & 1u);          // round-to-nearest-even
  return (unsigned short)(x >> 16);
}
__device__ __forceinline__ unsigned pk2bf(float lo, float hi) {
  return (unsigned)f2bf(lo) | ((unsigned)f2bf(hi) << 16);
}
__device__ __forceinline__ float bf2f(unsigned short h) {
  return __uint_as_float(((unsigned)h) << 16);
}
__device__ __forceinline__ float sigm(float x) { return 1.0f / (1.0f + __expf(-x)); }
__device__ __forceinline__ v8f vzero8() {
  v8f z;
#pragma unroll
  for (int e = 0; e < 8; ++e) z[e] = 0.0f;
  return z;
}

// ---------------------------------------------------------------------------
// Prep kernels
// ---------------------------------------------------------------------------
__global__ void k_zero(float* p, int n) {
  int i = blockIdx.x * blockDim.x + threadIdx.x;
  if (i < n) p[i] = 0.0f;
}

// Pack Wcat[dir][j][k] (bf16 pairs as dwords) and bias[dir][j] = b_ih+b_hh
__global__ void k_pack(const float* wihF, const float* whhF, const float* bihF, const float* bhhF,
                       const float* wihB, const float* whhB, const float* bihB, const float* bhhB,
                       unsigned* Wcat, float* bias) {
  int idx = blockIdx.x * blockDim.x + threadIdx.x;   // over 2*1024*256 dwords
  const int total = 2 * GATES * (KCAT / 2);
  if (idx < total) {
    int kd   = idx & 255;            // dword within row: halfs k=2kd, 2kd+1
    int dirj = idx >> 8;
    int j    = dirj & 1023;
    int dir  = dirj >> 10;
    const float* wih = dir ? wihB : wihF;
    const float* whh = dir ? whhB : whhF;
    int k0 = kd * 2, k1 = k0 + 1;
    float a = (k0 < D_SZ) ? wih[j * D_SZ + k0] : whh[j * D_SZ + (k0 - D_SZ)];
    float b = (k1 < D_SZ) ? wih[j * D_SZ + k1] : whh[j * D_SZ + (k1 - D_SZ)];
    Wcat[idx] = pk2bf(a, b);
  }
  if (idx < 2 * GATES) {
    int dir = idx >> 10, j = idx & 1023;
    bias[idx] = dir ? (bihB[j] + bhhB[j]) : (bihF[j] + bhhF[j]);
  }
}

// Segment-sum E[1:] rows into l2 buckets (blockDim = 256 = D)
__global__ void k_l2_accum(const float* E, const int* l2_of_leaf, float* sum, float* cnt) {
  for (int v = 1 + blockIdx.x; v < V_SZ; v += gridDim.x) {
    int seg = l2_of_leaf[v];
    atomicAdd(&sum[seg * D_SZ + threadIdx.x], E[(size_t)v * D_SZ + threadIdx.x]);
    if (threadIdx.x == 0) atomicAdd(&cnt[seg], 1.0f);
  }
}

__global__ void k_div(const float* sum, const float* cnt, float* out) {
  int r = blockIdx.x;
  out[r * D_SZ + threadIdx.x] = sum[r * D_SZ + threadIdx.x] / fmaxf(cnt[r], 1.0f);
}

__global__ void k_l3_accum(const float* level2, const int* l3_of_l2, float* sum, float* cnt) {
  for (int r = blockIdx.x; r < N2_SZ; r += gridDim.x) {
    int seg = l3_of_l2[r];
    atomicAdd(&sum[seg * D_SZ + threadIdx.x], level2[r * D_SZ + threadIdx.x]);
    if (threadIdx.x == 0) atomicAdd(&cnt[seg], 1.0f);
  }
}

__global__ void k_root(const float* level3, float* root) {
  float acc = 0.0f;
  for (int r = 0; r < N3_SZ; ++r) acc += level3[r * D_SZ + threadIdx.x];
  root[threadIdx.x] = acc * (1.0f / (float)N3_SZ);
}

// ---------------------------------------------------------------------------
// Fused bidirectional 4-step LSTM. 64 tokens per block, 256 threads (8 waves).
// LDS: [X(K=0..255) | H(K=256..511)] per row, bf16 -> 64KB.
// Each wave owns col-blocks {2w, 2w+1} x row-blocks 0..3; K=512 via
// v_wmma_f32_16x16x32_bf16; 4 gate accumulators per 16x16 tile; c in regs.
// ---------------------------------------------------------------------------
__global__ __launch_bounds__(256) void k_lstm_fused(
    const int* __restrict__ tok, const int* __restrict__ l2_of_leaf,
    const int* __restrict__ l3_of_l2,
    const float* __restrict__ E, const float* __restrict__ level2,
    const float* __restrict__ level3, const float* __restrict__ root,
    const unsigned* __restrict__ Wcat, const float* __restrict__ bias,
    float* __restrict__ out) {
  __shared__ unsigned short sXH[64 * KCAT];   // 65536 bytes

  const int tid  = threadIdx.x;
  const int lane = tid & 31;
  const int wave = tid >> 5;
  const int m16  = lane & 15;
  const int hsel = (lane >> 4) & 1;
  const int gbase = blockIdx.x * 64;

  // Gather assignment: thread covers row (tid>>2), 64 cols at (tid&3)*64
  const int grow  = tid >> 2;
  const int cbase = (tid & 3) * 64;
  const int v   = tok[gbase + grow];
  const int i2  = l2_of_leaf[v];
  const int i3  = l3_of_l2[i2];
  const float msk = (v != 0) ? 1.0f : 0.0f;

  for (int dir = 0; dir < 2; ++dir) {
    const unsigned* W  = Wcat + (size_t)dir * GATES * (KCAT / 2);
    const float*    bb = bias + dir * GATES;

    // reset h (LDS) and cell state (regs)
#pragma unroll 4
    for (int i = 0; i < 64; i += 2)
      *(unsigned*)&sXH[grow * KCAT + D_SZ + cbase + i] = 0u;
    v8f cst[2][4];
#pragma unroll
    for (int a = 0; a < 2; ++a)
#pragma unroll
      for (int b = 0; b < 4; ++b) cst[a][b] = vzero8();
    __syncthreads();

    for (int t = 0; t < 4; ++t) {
      const int lvl = dir ? (3 - t) : t;
      // ---- gather x_t (masked) into LDS as bf16 ----
      const float* src;
      if      (lvl == 0) src = E + (size_t)v * D_SZ;
      else if (lvl == 1) src = level2 + i2 * D_SZ;
      else if (lvl == 2) src = level3 + i3 * D_SZ;
      else               src = root;
      const float4* s4 = (const float4*)(src + cbase);
#pragma unroll 4
      for (int i = 0; i < 16; ++i) {
        float4 x = s4[i];
        *(unsigned*)&sXH[grow * KCAT + cbase + i * 4]     = pk2bf(x.x * msk, x.y * msk);
        *(unsigned*)&sXH[grow * KCAT + cbase + i * 4 + 2] = pk2bf(x.z * msk, x.w * msk);
      }
      __syncthreads();

      // ---- z = [X|H] @ Wcat^T + b ; gate math ; h in regs ----
      v8f hnew[2][4];
#pragma unroll
      for (int cbi = 0; cbi < 2; ++cbi) {
        const int cb = wave * 2 + cbi;
        v8f acc[4][4];
#pragma unroll
        for (int rb = 0; rb < 4; ++rb)
#pragma unroll
          for (int g = 0; g < 4; ++g) acc[rb][g] = vzero8();

        for (int kc = 0; kc < 16; ++kc) {
          // B fragments (one per gate), streamed from L2-resident Wcat
          FragBF bfr[4];
#pragma unroll
          for (int g = 0; g < 4; ++g) {
            const unsigned* wrow = W + (size_t)(g * D_SZ + cb * 16 + m16) * (KCAT / 2);
#pragma unroll
            for (int vv = 0; vv < 8; ++vv)
              bfr[g].u[vv] = wrow[kc * 16 + vv + 8 * hsel];
          }
#pragma unroll
          for (int rb = 0; rb < 4; ++rb) {
            FragBF a;
            const int row = rb * 16 + m16;
#pragma unroll
            for (int vv = 0; vv < 8; ++vv) {
              const int k0 = 2 * vv + 16 * hsel;
              a.u[vv] = *(const unsigned*)&sXH[row * KCAT + kc * 32 + k0];
            }
#pragma unroll
            for (int g = 0; g < 4; ++g)
              acc[rb][g] = __builtin_amdgcn_wmma_f32_16x16x32_bf16(
                  false, a.bf, false, bfr[g].bf, (short)0, acc[rb][g], false, false);
          }
        }

        const float bi = bb[0 * D_SZ + cb * 16 + m16];
        const float bf_ = bb[1 * D_SZ + cb * 16 + m16];
        const float bg = bb[2 * D_SZ + cb * 16 + m16];
        const float bo = bb[3 * D_SZ + cb * 16 + m16];
#pragma unroll
        for (int rb = 0; rb < 4; ++rb) {
#pragma unroll
          for (int e = 0; e < 8; ++e) {
            const float ig = sigm(acc[rb][0][e] + bi);
            const float fg = sigm(acc[rb][1][e] + bf_);
            const float gg = tanhf(acc[rb][2][e] + bg);
            const float og = sigm(acc[rb][3][e] + bo);
            const float c  = fg * cst[cbi][rb][e] + ig * gg;
            cst[cbi][rb][e] = c;
            hnew[cbi][rb][e] = og * tanhf(c);
          }
        }
      }
      __syncthreads();   // all waves done reading H for this step

      // write new h to LDS (C/D layout: row = rb*16 + e + 8*hsel, col = lane%16)
#pragma unroll
      for (int cbi = 0; cbi < 2; ++cbi) {
        const int col = D_SZ + (wave * 2 + cbi) * 16 + m16;
#pragma unroll
        for (int rb = 0; rb < 4; ++rb)
#pragma unroll
          for (int e = 0; e < 8; ++e)
            sXH[(rb * 16 + e + 8 * hsel) * KCAT + col] = f2bf(hnew[cbi][rb][e]);
      }
      __syncthreads();
    }

    // final hidden -> out[:, dir*256 : dir*256+256]
#pragma unroll 4
    for (int i = 0; i < 64; ++i)
      out[(size_t)(gbase + grow) * (2 * D_SZ) + dir * D_SZ + cbase + i] =
          bf2f(sXH[grow * KCAT + D_SZ + cbase + i]);
    __syncthreads();
  }
}

// ---------------------------------------------------------------------------
extern "C" void kernel_launch(void* const* d_in, const int* in_sizes, int n_in,
                              void* d_out, int out_size, void* d_ws, size_t ws_size,
                              hipStream_t stream) {
  const int*   tok        = (const int*)d_in[0];
  const int*   l2_of_leaf = (const int*)d_in[1];
  const int*   l3_of_l2   = (const int*)d_in[2];
  const float* embed      = (const float*)d_in[3];
  const float* wihF = (const float*)d_in[4];
  const float* whhF = (const float*)d_in[5];
  const float* bihF = (const float*)d_in[6];
  const float* bhhF = (const float*)d_in[7];
  const float* wihB = (const float*)d_in[8];
  const float* whhB = (const float*)d_in[9];
  const float* bihB = (const float*)d_in[10];
  const float* bhhB = (const float*)d_in[11];
  float* out = (float*)d_out;

  // Workspace layout (floats, then dwords for bf16 weights) ~4.3 MB total
  float* ws     = (float*)d_ws;
  float* l2sum  = ws;                       // N2*D = 256000
  float* l2cnt  = l2sum + N2_SZ * D_SZ;     // 1024 (padded)
  float* l3sum  = l2cnt + 1024;             // N3*D = 25600
  float* l3cnt  = l3sum + N3_SZ * D_SZ;     // 256 (padded)
  float* level2 = l3cnt + 256;              // 256000
  float* level3 = level2 + N2_SZ * D_SZ;    // 25600
  float* root   = level3 + N3_SZ * D_SZ;    // 256
  float* bias   = root + D_SZ;              // 2048
  unsigned* Wcat = (unsigned*)(bias + 2 * GATES);  // 2*1024*256 dwords = 2MB

  const int nzero = N2_SZ * D_SZ + 1024 + N3_SZ * D_SZ + 256;  // sums + counts
  k_zero<<<(nzero + 255) / 256, 256, 0, stream>>>(ws, nzero);

  k_pack<<<(2 * GATES * (KCAT / 2) + 255) / 256, 256, 0, stream>>>(
      wihF, whhF, bihF, bhhF, wihB, whhB, bihB, bhhB, Wcat, bias);

  k_l2_accum<<<1024, 256, 0, stream>>>(embed, l2_of_leaf, l2sum, l2cnt);
  k_div<<<N2_SZ, 256, 0, stream>>>(l2sum, l2cnt, level2);
  k_l3_accum<<<256, 256, 0, stream>>>(level2, l3_of_l2, l3sum, l3cnt);
  k_div<<<N3_SZ, 256, 0, stream>>>(l3sum, l3cnt, level3);
  k_root<<<1, 256, 0, stream>>>(level3, root);

  k_lstm_fused<<<NTOK / 64, 256, 0, stream>>>(
      tok, l2_of_leaf, l3_of_l2, embed, level2, level3, root, Wcat, bias, out);
}
